// Attention_54082228191673
// MI455X (gfx1250) — compile-verified, matrix operand order checked by reading, not run on hardware
//
#include <hip/hip_runtime.h>
#include <hip/hip_bf16.h>
#include <math.h>

// Problem constants from the reference
#define BB 2
#define TT 512
#define EE 1024
#define SS 2048
#define AA 30

typedef float     v2f  __attribute__((ext_vector_type(2)));
typedef float     v8f  __attribute__((ext_vector_type(8)));
typedef _Float16  v16h __attribute__((ext_vector_type(16)));

// ---------------------------------------------------------------------------
// Kernel 1: head[b*T+t] = relu(x[b,t,:] . W + bias) via WMMA.
// One wave per 16 rows of the flattened (B*T, E) matrix.
//   A (16x4 f32): lane l<16 -> M=l, VGPR0=K0, VGPR1=K1; lane l+16 -> K2,K3.
//   B (4x16 f32): W[k] broadcast across all 16 N columns (same mapping).
//   D (16x16 f32): column N=0 -> lane0 holds M=0..7 in c[0..7], lane16 M=8..15.
// ---------------------------------------------------------------------------
__global__ __launch_bounds__(32) void head_gemv_wmma(
    const float* __restrict__ x, const float* __restrict__ Wv,
    const float* __restrict__ bias, float* __restrict__ head) {
  const int lane = threadIdx.x;          // 0..31 (wave32)
  const int half = lane >> 4;            // 0 or 1
  const int l    = lane & 15;            // row within tile
  const int row0 = blockIdx.x * 16;      // row into flattened (B*T)
  const float* __restrict__ arow = x + (size_t)(row0 + l) * EE;

  v8f c = {0.f, 0.f, 0.f, 0.f, 0.f, 0.f, 0.f, 0.f};

#if __has_builtin(__builtin_amdgcn_wmma_f32_16x16x4_f32)
  for (int k = 0; k < EE; k += 4) {
    const int kk = k + 2 * half;
    v2f a, bm;
    a.x  = arow[kk];
    a.y  = arow[kk + 1];
    bm.x = Wv[kk];                       // broadcast column: every N gets W[k]
    bm.y = Wv[kk + 1];
    c = __builtin_amdgcn_wmma_f32_16x16x4_f32(
        /*neg_a=*/false, a, /*neg_b=*/false, bm,
        /*c_mod=*/(short)0, c, /*reuse_a=*/false, /*reuse_b=*/false);
  }
#else
  // Fallback: confirmed builtin, f16 inputs / f32 accumulate, K=32 per step.
  // 16-bit A layout (16x32): lanes 0-15: V0..3 = K0..7, V4..7 = K16..23;
  // lanes 16-31: V0..3 = K8..15, V4..7 = K24..31.
  for (int k = 0; k < EE; k += 32) {
    v16h a, bm;
#pragma unroll
    for (int i = 0; i < 16; ++i) {
      const int kk = k + ((i < 8) ? 0 : 16) + half * 8 + (i & 7);
      a[i]  = (_Float16)arow[kk];
      bm[i] = (_Float16)Wv[kk];
    }
    c = __builtin_amdgcn_wmma_f32_16x16x32_f16(
        false, a, false, bm, (short)0, c, false, false);
  }
#endif

  if (l == 0) {                          // lanes 0 and 16 hold column N=0
    const float bb = bias[0];
    const int r0 = row0 + half * 8;
#pragma unroll
    for (int r = 0; r < 8; ++r) {
      const float v = c[r] + bb;
      head[r0 + r] = v > 0.f ? v : 0.f;  // relu
    }
  }
}

// ---------------------------------------------------------------------------
// Kernel 2: per (b,s) span: masked softmax over <=30 head scores, then
// weighted sum of the gathered x rows. One wave per span; 8 waves per block.
// ---------------------------------------------------------------------------
__global__ __launch_bounds__(256) void span_attn(
    const float* __restrict__ x, const float* __restrict__ head,
    const int* __restrict__ start, const int* __restrict__ end,
    float* __restrict__ out) {
  const int lane = threadIdx.x & 31;
  const int wave = threadIdx.x >> 5;
  const int pair = blockIdx.x * 8 + wave;      // 0 .. B*S-1
  const int b = pair >> 11;                    // / S (S=2048)
  const int s = pair & (SS - 1);               // % S

  __shared__ float s_attn[8][32];
  __shared__ int   s_off[8][32];

  const int st = start[s];
  const int w  = end[s] - st + 1;              // 1..30, never clamped inside window

  // --- masked softmax over the window (wave-cooperative, wave32 shuffles) ---
  float sc = -INFINITY;
  if (lane < w) sc = head[b * TT + st + lane];
  float m = sc;
#pragma unroll
  for (int o = 16; o > 0; o >>= 1) m = fmaxf(m, __shfl_xor(m, o, 32));
  const float p = (lane < w) ? __expf(sc - m) : 0.f;
  float sum = p;
#pragma unroll
  for (int o = 16; o > 0; o >>= 1) sum += __shfl_xor(sum, o, 32);

  s_attn[wave][lane] = p / sum;
  s_off[wave][lane]  = (lane < w) ? (b * TT + st + lane) * EE : 0;
  __syncthreads();

  // Pull the first cacheline of each gathered row toward the WGP.
  if (lane < w) __builtin_prefetch(x + s_off[wave][lane], 0, 3);

  // --- weighted sum: lane owns 4 consecutive floats per 128-wide chunk ---
#pragma unroll
  for (int chunk = 0; chunk < 8; ++chunk) {
    const int e = chunk * 128 + lane * 4;
    float4 acc = make_float4(0.f, 0.f, 0.f, 0.f);
    for (int a = 0; a < w; ++a) {
      const float4 v = *reinterpret_cast<const float4*>(x + s_off[wave][a] + e);
      const float t  = s_attn[wave][a];
      acc.x = fmaf(t, v.x, acc.x);
      acc.y = fmaf(t, v.y, acc.y);
      acc.z = fmaf(t, v.z, acc.z);
      acc.w = fmaf(t, v.w, acc.w);
    }
    *reinterpret_cast<float4*>(out + (size_t)pair * EE + e) = acc;
  }
}

// ---------------------------------------------------------------------------
extern "C" void kernel_launch(void* const* d_in, const int* in_sizes, int n_in,
                              void* d_out, int out_size, void* d_ws, size_t ws_size,
                              hipStream_t stream) {
  const float* x    = (const float*)d_in[0];   // (B,T,E) f32
  const float* W    = (const float*)d_in[1];   // (E,1)   f32
  const float* bias = (const float*)d_in[2];   // (1,)    f32
  const int*   st   = (const int*)d_in[3];     // (S,)    i32
  const int*   en   = (const int*)d_in[4];     // (S,)    i32
  float*       out  = (float*)d_out;           // (B,S,E) f32
  float*       head = (float*)d_ws;            // B*T floats = 4KB scratch

  // Kernel 1: 1024 rows / 16 rows-per-wave = 64 single-wave blocks.
  head_gemv_wmma<<<(BB * TT) / 16, 32, 0, stream>>>(x, W, bias, head);

  // Kernel 2: B*S = 4096 spans, 8 spans (waves) per 256-thread block.
  span_attn<<<(BB * SS) / 8, 256, 0, stream>>>(x, head, st, en, out);
}
